// HelmholtzLoss_13219909337725
// MI455X (gfx1250) — matrix operand order repaired and checked by reading, not compiled, not run on hardware
//
#include <hip/hip_runtime.h>
#include <hip/hip_bf16.h>

// ---- types -----------------------------------------------------------------
typedef __attribute__((ext_vector_type(16))) _Float16 v16h;
typedef __attribute__((ext_vector_type(8)))  _Float16 v8h;
typedef __attribute__((ext_vector_type(8)))  float    v8f;

#define HIDDEN 256
#define HPAD   264              // 256 + 8 halfs pad -> row stride 528B (bank spread)
#define TPB    128              // 4 waves, one 16-point tile per workgroup

// Hardware tanh if the toolchain exposes it (gfx1250 has V_TANH_F32), else OCML.
__device__ __forceinline__ float fast_tanh(float x) {
#if __has_builtin(__builtin_amdgcn_tanhf)
    return __builtin_amdgcn_tanhf(x);
#elif __has_builtin(__builtin_amdgcn_tanh_f32)
    return __builtin_amdgcn_tanh_f32(x);
#else
    return tanhf(x);
#endif
}

__device__ __forceinline__ v16h cat16(v8h lo, v8h hi) {
    return __builtin_shufflevector(lo, hi, 0,1,2,3,4,5,6,7,8,9,10,11,12,13,14,15);
}
__device__ __forceinline__ v8f zero8() {
    v8f z = {0.f,0.f,0.f,0.f,0.f,0.f,0.f,0.f};
    return z;
}

// ---- weight pre-swizzle: W[k][n] (row-major 256x256 f32) -> f16 A-fragment order
// A 16x32 f16 fragment, lane L: M = L%16, elem j: K = (L/16)*16 + j.
// Fragment stream: [ntile(16)][kchunk(8)][lane(32)][j(16)] contiguous halfs.
__global__ void helm_prep_weights(const float* __restrict__ W2,
                                  const float* __restrict__ W3,
                                  _Float16* __restrict__ f2,
                                  _Float16* __restrict__ f3) {
    int e = blockIdx.x * blockDim.x + threadIdx.x;        // 0 .. 131071
    const float* W = (e < 65536) ? W2 : W3;
    _Float16*    f = (e < 65536) ? f2 : f3;
    int i    = e & 65535;
    int j    = i & 15;
    int lane = (i >> 4) & 31;
    int kc   = (i >> 9) & 7;
    int nt   = (i >> 12) & 15;
    int n = nt * 16 + (lane & 15);
    int k = kc * 32 + ((lane >> 4) << 4) + j;
    f[i] = (_Float16)W[k * HIDDEN + n];
}

// ---- one hidden layer: z/dz0/dz1/dz2/lz GEMMs + tanh chain rule ------------
// act[q][m][k]: q = {0:a, 1:g0, 2:g1, 3:g2, 4:lap}, m = point in tile, k = neuron.
__device__ __forceinline__ void hidden_layer(const _Float16* __restrict__ frag,
                                             const float*    __restrict__ bias,
                                             _Float16 (*act)[16][HPAD],
                                             int wave, int lane) {
    const int m    = lane & 15;
    const int half = lane >> 4;
    v8h pk[4][5];                                 // layer outputs staged in VGPRs

    #pragma unroll
    for (int tt = 0; tt < 4; ++tt) {
        const int ntile = wave * 4 + tt;
        v8f acc[5];
        #pragma unroll
        for (int q = 0; q < 5; ++q) acc[q] = zero8();

        #pragma unroll
        for (int c = 0; c < 8; ++c) {
            // A = weight fragment (broadcast across waves; L2-resident)
            const v8h* wp = (const v8h*)(frag + ((size_t)((ntile * 8 + c) * 32 + lane)) * 16);
            v16h aF = cat16(wp[0], wp[1]);
            // B = activation fragment: lane col = point m, 16 contiguous k
            #pragma unroll
            for (int q = 0; q < 5; ++q) {
                const v8h* bp = (const v8h*)(&act[q][m][c * 32 + half * 16]);
                v16h bF = cat16(bp[0], bp[1]);
                acc[q] = __builtin_amdgcn_wmma_f32_16x16x32_f16(
                    false, aF, false, bF, (short)0, acc[q], false, false);
            }
        }

        // pointwise tanh + chain rule (f32), pack to f16
        const int nbase = ntile * 16 + 8 * half;  // D: reg r -> neuron nbase + r
        #pragma unroll
        for (int r = 0; r < 8; ++r) {
            float z  = acc[0][r] + bias[nbase + r];
            float t  = fast_tanh(z);
            float s  = 1.0f - t * t;
            float d0 = acc[1][r], d1 = acc[2][r], d2 = acc[3][r];
            pk[tt][0][r] = (_Float16)t;
            pk[tt][1][r] = (_Float16)(s * d0);
            pk[tt][2][r] = (_Float16)(s * d1);
            pk[tt][3][r] = (_Float16)(s * d2);
            pk[tt][4][r] = (_Float16)(s * acc[4][r] - 2.0f * t * s * (d0*d0 + d1*d1 + d2*d2));
        }
    }

    __syncthreads();   // all waves finished READING act before overwriting it
    #pragma unroll
    for (int tt = 0; tt < 4; ++tt) {
        const int ntile = wave * 4 + tt;
        #pragma unroll
        for (int q = 0; q < 5; ++q)
            *(v8h*)(&act[q][m][ntile * 16 + half * 8]) = pk[tt][q];   // ds_store_b128
    }
    __syncthreads();   // writes visible to next layer
}

// ---- main kernel: 16 points per workgroup ----------------------------------
__global__ __launch_bounds__(TPB)
void helm_main(const float* __restrict__ inputs,
               const float* __restrict__ W1, const float* __restrict__ b1,
               const _Float16* __restrict__ f2, const float* __restrict__ b2,
               const _Float16* __restrict__ f3, const float* __restrict__ b3,
               const float* __restrict__ W4, const float* __restrict__ b4,
               float* __restrict__ partial) {
    __shared__ __align__(16) _Float16 act[5][16][HPAD];   // ~42 KB
    __shared__ float redU[TPB], redL[TPB], red2[16];

    const float K2C = (343.0f / (2.0f * 3.14159265358979323846f * 1000.0f)) *
                      (343.0f / (2.0f * 3.14159265358979323846f * 1000.0f));

    const int tid  = threadIdx.x;
    const int wave = tid >> 5;
    const int lane = tid & 31;
    const int base = blockIdx.x * 16;

    // ---- layer 1 (3 -> 256), exact chain-rule seed, pure VALU --------------
    {
        const int m   = tid >> 3;     // point 0..15
        const int seg = tid & 7;      // 32 neurons per thread
        float x  = inputs[(base + m) * 3 + 0];
        float y  = inputs[(base + m) * 3 + 1];
        float zc = inputs[(base + m) * 3 + 2];
        #pragma unroll
        for (int nn = 0; nn < 32; nn += 8) {
            v8h pa, pg0, pg1, pg2, pl;
            #pragma unroll
            for (int r = 0; r < 8; ++r) {
                int n = seg * 32 + nn + r;
                float w0 = W1[n], w1 = W1[HIDDEN + n], w2 = W1[2 * HIDDEN + n];
                float z = b1[n] + x * w0 + y * w1 + zc * w2;
                float t = fast_tanh(z);
                float s = 1.0f - t * t;
                pa[r]  = (_Float16)t;
                pg0[r] = (_Float16)(s * w0);
                pg1[r] = (_Float16)(s * w1);
                pg2[r] = (_Float16)(s * w2);
                pl[r]  = (_Float16)(-2.0f * t * s * (w0*w0 + w1*w1 + w2*w2));
            }
            int kk = seg * 32 + nn;
            *(v8h*)(&act[0][m][kk]) = pa;
            *(v8h*)(&act[1][m][kk]) = pg0;
            *(v8h*)(&act[2][m][kk]) = pg1;
            *(v8h*)(&act[3][m][kk]) = pg2;
            *(v8h*)(&act[4][m][kk]) = pl;
        }
    }
    __syncthreads();

    // ---- two 256x256 hidden layers via WMMA --------------------------------
    hidden_layer(f2, b2, act, wave, lane);
    hidden_layer(f3, b3, act, wave, lane);

    // ---- final layer (256 -> 1): u and lap dot products, then residual -----
    {
        const int m   = tid & 15;
        const int seg = tid >> 4;
        float su = 0.f, sl = 0.f;
        #pragma unroll 4
        for (int k = seg * 32; k < seg * 32 + 32; ++k) {
            float w = W4[k];
            su += w * (float)act[0][m][k];
            sl += w * (float)act[4][m][k];
        }
        redU[tid] = su;
        redL[tid] = sl;
    }
    __syncthreads();
    if (tid < 16) {
        float u = b4[0], l = 0.f;
        #pragma unroll
        for (int s = 0; s < 8; ++s) { u += redU[tid + 16 * s]; l += redL[tid + 16 * s]; }
        float res = l + K2C * u;
        red2[tid] = res * res;
    }
    __syncthreads();
    if (tid == 0) {
        float a = 0.f;
        #pragma unroll
        for (int i = 0; i < 16; ++i) a += red2[i];
        partial[blockIdx.x] = a;
    }
}

// ---- final reduction (deterministic, no atomics) ---------------------------
__global__ __launch_bounds__(256)
void helm_reduce(const float* __restrict__ partial, float* __restrict__ out,
                 int nblocks, float inv_n) {
    __shared__ float s[256];
    float acc = 0.f;
    for (int i = threadIdx.x; i < nblocks; i += 256) acc += partial[i];
    s[threadIdx.x] = acc;
    __syncthreads();
    for (int off = 128; off > 0; off >>= 1) {
        if (threadIdx.x < off) s[threadIdx.x] += s[threadIdx.x + off];
        __syncthreads();
    }
    if (threadIdx.x == 0) out[0] = s[0] * inv_n;
}

// ---- launch ----------------------------------------------------------------
extern "C" void kernel_launch(void* const* d_in, const int* in_sizes, int n_in,
                              void* d_out, int out_size, void* d_ws, size_t ws_size,
                              hipStream_t stream) {
    const float* inputs = (const float*)d_in[0];
    const float* W1 = (const float*)d_in[1];
    const float* b1 = (const float*)d_in[2];
    const float* W2 = (const float*)d_in[3];
    const float* b2 = (const float*)d_in[4];
    const float* W3 = (const float*)d_in[5];
    const float* b3 = (const float*)d_in[6];
    const float* W4 = (const float*)d_in[7];
    const float* b4 = (const float*)d_in[8];

    const int npts    = in_sizes[0] / 3;          // 262144
    const int nblocks = npts / 16;                // 16384

    // workspace layout: f16 W2 frags (128KB) | f16 W3 frags (128KB) | partials (64KB)
    _Float16* f2 = (_Float16*)d_ws;
    _Float16* f3 = f2 + (HIDDEN * HIDDEN);
    float* partial = (float*)((char*)d_ws + 2u * HIDDEN * HIDDEN * sizeof(_Float16));

    helm_prep_weights<<<(2 * HIDDEN * HIDDEN) / 256, 256, 0, stream>>>(W2, W3, f2, f3);
    helm_main<<<nblocks, TPB, 0, stream>>>(inputs, W1, b1, f2, b2, f3, b3, W4, b4, partial);
    helm_reduce<<<1, 256, 0, stream>>>(partial, (float*)d_out, nblocks, 1.0f / (float)npts);
}
